// MySelfAttention_V2_73057393705742
// MI455X (gfx1250) — matrix operand order repaired and checked by reading, not compiled, not run on hardware
//
#include <hip/hip_runtime.h>

// ---------------- problem constants (from reference) ----------------
constexpr int BB   = 4;
constexpr int NN   = 4096;
constexpr int DD   = 768;
constexpr int D3   = 3 * DD;      // 2304
constexpr int MTOT = BB * NN;     // 16384
constexpr float SCALE = 0.03608439182435161f;   // 1/sqrt(768)

// ---------------- WMMA types (CDNA5, wave32) ----------------
typedef __bf16 bf16x16 __attribute__((ext_vector_type(16)));
typedef float  f32x8   __attribute__((ext_vector_type(8)));

union Frag16 {
  bf16x16        v;
  unsigned short u[16];
};

__device__ __forceinline__ unsigned short f32_to_bf16(float f) {
  unsigned int x = __float_as_uint(f);
  x += 0x7fffu + ((x >> 16) & 1u);       // round-to-nearest-even
  return (unsigned short)(x >> 16);
}

__device__ __forceinline__ f32x8 zero8() {
  f32x8 z = {0.f, 0.f, 0.f, 0.f, 0.f, 0.f, 0.f, 0.f};
  return z;
}

__device__ __forceinline__ f32x8 wmma_bf16(const Frag16& a, const Frag16& b, f32x8 c) {
  return __builtin_amdgcn_wmma_f32_16x16x32_bf16(false, a.v, false, b.v,
                                                 (short)0, c, false, false);
}

// ---------------- CDNA5 async global->LDS copy (ASYNCcnt path) ----------
__device__ __forceinline__ void async_b128(const void* lds_ptr, const void* gptr) {
  unsigned loff = (unsigned)(unsigned long long)lds_ptr;   // LDS byte offset
  asm volatile("global_load_async_to_lds_b128 %0, %1, off"
               :: "v"(loff), "v"(gptr)
               : "memory");
}
__device__ __forceinline__ void wait_async() {
  asm volatile("s_wait_asynccnt 0" ::: "memory");
}

// Wave32 fragment layouts (CDNA5 ISA 7.12.2):
//  A 16x32 bf16 : lane->M=lane&15 ; half=lane>>4 ; vgpr i -> K base
//                 kb = (i<4 ? 0 : 16) + half*8 + (i&3)*2  (pair kb,kb+1)
//  B 32x16 bf16 : lane->N=lane&15 ; K order as A; code packs A and B as K = half*16 + 2*i (pair)
//  C 16x16 f32  : lane->N=lane&15 ; vgpr r -> M = half*8 + r

constexpr int TSTR = 40;   // 128x32 tile row stride (80B rows, 16B aligned)

// A-fragment gather from a row-major [row][k] LDS tile
__device__ __forceinline__ void load_afrag(Frag16& a, const unsigned short* base,
                                           int row, int stride, int kofs, int half) {
  #pragma unroll
  for (int i = 0; i < 8; ++i) {
    int kb = half * 16 + 2 * i;
    a.u[2*i]   = base[row * stride + kofs + kb];
    a.u[2*i+1] = base[row * stride + kofs + kb + 1];
  }
}
// B-fragment gather from a [n][k] (k contiguous) LDS tile
__device__ __forceinline__ void load_bfrag_nk(Frag16& b, const unsigned short* base,
                                              int n, int stride, int half) {
  #pragma unroll
  for (int i = 0; i < 8; ++i) {
    int kk = half * 16 + 2 * i;
    b.u[2*i]   = base[n * stride + kk];
    b.u[2*i+1] = base[n * stride + kk + 1];
  }
}

// =====================================================================
// Kernel 1: QKV projection  C[16384 x 2304] = X(f32)*Wqkv(f32) + b
// (f32 inputs -> convert while staging; VGPR path)
// =====================================================================
__global__ __launch_bounds__(256)
void qkv_proj_kernel(const float* __restrict__ x, const float* __restrict__ w,
                     const float* __restrict__ bias,
                     unsigned short* __restrict__ qo,
                     unsigned short* __restrict__ ko,
                     unsigned short* __restrict__ vo)
{
  __shared__ unsigned short As[128 * TSTR];   // [m][k]
  __shared__ unsigned short Bs[128 * TSTR];   // [n][k] (transposed W tile)

  const int m0   = blockIdx.x * 128;
  const int n0   = blockIdx.y * 128;
  const int tid  = threadIdx.x;
  const int wave = tid >> 5;
  const int lane = tid & 31;
  const int half = lane >> 4;
  const int ln   = lane & 15;

  f32x8 acc[8];
  #pragma unroll
  for (int f = 0; f < 8; ++f) acc[f] = zero8();

  for (int k0 = 0; k0 < DD; k0 += 32) {
    __syncthreads();
    #pragma unroll 4
    for (int i = tid; i < 128 * 32; i += 256) {
      int r = i >> 5, c = i & 31;
      As[r * TSTR + c] = f32_to_bf16(x[(size_t)(m0 + r) * DD + k0 + c]);
    }
    #pragma unroll 4
    for (int i = tid; i < 32 * 128; i += 256) {
      int kk = i >> 7, nn = i & 127;
      Bs[nn * TSTR + kk] = f32_to_bf16(w[(size_t)(k0 + kk) * D3 + n0 + nn]);
    }
    __syncthreads();

    Frag16 a, b0, b1;
    load_afrag(a, As, wave * 16 + ln, TSTR, 0, half);
    load_bfrag_nk(b0, Bs, 0 * 16 + ln, TSTR, half);
    #pragma unroll
    for (int f = 0; f < 8; f += 2) {                 // B double-buffered
      load_bfrag_nk(b1, Bs, (f + 1) * 16 + ln, TSTR, half);
      acc[f] = wmma_bf16(a, b0, acc[f]);
      if (f + 2 < 8) load_bfrag_nk(b0, Bs, (f + 2) * 16 + ln, TSTR, half);
      acc[f + 1] = wmma_bf16(a, b1, acc[f + 1]);
    }
  }

  #pragma unroll
  for (int f = 0; f < 8; ++f) {
    const int   e  = n0 + f * 16 + ln;
    const float bv = bias[e];
    #pragma unroll
    for (int r = 0; r < 8; ++r) {
      const int grow = m0 + wave * 16 + half * 8 + r;
      const unsigned short h = f32_to_bf16(acc[f][r] + bv);
      if      (e <     DD) qo[(size_t)grow * DD +  e          ] = h;
      else if (e < 2 * DD) ko[(size_t)grow * DD + (e -     DD)] = h;
      else                 vo[(size_t)grow * DD + (e - 2 * DD)] = h;
    }
  }
}

// =====================================================================
// Kernel 2: flash attention. One block = 32 query rows, 8 waves.
//  - QK^T: wave w -> keys [16w,16w+16), 2 row groups; K slice DMA is
//    double-buffered (async fill of slice d0+32 overlaps WMMA on d0)
//  - online softmax in LDS (32 rows, one per thread of wave 0)
//  - P*V : waves split D (96 cols each) x 2 row groups
// =====================================================================
constexpr int QS_STRIDE = 776;   // 1552B rows, 16B aligned
constexpr int PB_STRIDE = 136;
constexpr int KS_STRIDE = 40;    // 80B rows, 16B aligned
constexpr int VS_STRIDE = 776;
constexpr int SB_STRIDE = 130;   // f32; 130 dwords -> conflict-free row scans

constexpr int OFF_Q  = 0;                           // 32*776*2 = 49664
constexpr int OFF_P  = OFF_Q + 32 * QS_STRIDE * 2;  // 32*136*2 =  8704
constexpr int OFF_ST = OFF_P + 32 * PB_STRIDE * 2;  // 96 floats =  384
constexpr int OFF_KS = OFF_ST + 96 * 4;             // max(2*K, Sb) = 20480
constexpr int KB_BYTES = 128 * KS_STRIDE * 2;       // 10240 per K buffer
constexpr int OFF_V  = OFF_KS + 2 * KB_BYTES;       // 32*776*2 = 49664
constexpr int ATT_SMEM = OFF_V + 32 * VS_STRIDE * 2; // 128896 bytes

__global__ __launch_bounds__(256)
void flash_attn_kernel(const unsigned short* __restrict__ qg,
                       const unsigned short* __restrict__ kg,
                       const unsigned short* __restrict__ vg,
                       unsigned short* __restrict__ ctx)
{
  extern __shared__ char smem[];
  unsigned short* Qs  = (unsigned short*)(smem + OFF_Q);
  unsigned short* Pb  = (unsigned short*)(smem + OFF_P);
  float*          m_i = (float*)(smem + OFF_ST);
  float*          l_i = m_i + 32;
  float*          rsc = m_i + 64;
  unsigned short* Kb0 = (unsigned short*)(smem + OFF_KS);
  unsigned short* Kb1 = (unsigned short*)(smem + OFF_KS + KB_BYTES);
  float*          Sb  = (float*)(smem + OFF_KS);          // aliases K buffers
  unsigned short* Vs  = (unsigned short*)(smem + OFF_V);

  const int tid  = threadIdx.x;
  const int wave = tid >> 5;
  const int lane = tid & 31;
  const int half = lane >> 4;
  const int ln   = lane & 15;

  const int    qblk  = blockIdx.x;
  const int    batch = qblk / (NN / 32);
  const size_t qrow0 = (size_t)qblk * 32;
  const size_t krow0 = (size_t)batch * NN;

  // ---- async-fill the Q tile: 32 rows x 768 bf16 = 32*96 16B chunks ----
  #pragma unroll
  for (int it = 0; it < 12; ++it) {
    int id = tid + it * 256;
    int r = id / 96, c = id - r * 96;
    async_b128(Qs + r * QS_STRIDE + c * 8, qg + (qrow0 + r) * DD + c * 8);
  }
  if (tid < 32) { m_i[tid] = -1e30f; l_i[tid] = 0.f; }

  f32x8 acc0[6], acc1[6];
  #pragma unroll
  for (int f = 0; f < 6; ++f) { acc0[f] = zero8(); acc1[f] = zero8(); }
  const int dbase = wave * 96;

  for (int j = 0; j < NN; j += 128) {
    // ---- S = Q.K^T ; K slice DMA double-buffered ----
    f32x8 sacc0 = zero8(), sacc1 = zero8();
    #pragma unroll
    for (int it = 0; it < 2; ++it) {              // prologue: slice d0=0
      int id = tid + it * 256;
      int r = id >> 2, c = id & 3;
      async_b128(Kb0 + r * KS_STRIDE + c * 8,
                 kg + (krow0 + j + r) * DD + c * 8);
    }
    wait_async();
    __syncthreads();
    for (int itd = 0; itd < DD / 32; ++itd) {
      const int d0 = itd * 32;
      const unsigned short* kcur = (itd & 1) ? Kb1 : Kb0;
      if (itd + 1 < DD / 32) {
        unsigned short* knxt = (itd & 1) ? Kb0 : Kb1;
        #pragma unroll
        for (int it = 0; it < 2; ++it) {
          int id = tid + it * 256;
          int r = id >> 2, c = id & 3;
          async_b128(knxt + r * KS_STRIDE + c * 8,
                     kg + (krow0 + j + r) * DD + (d0 + 32) + c * 8);
        }
      }
      Frag16 a0, a1, b;
      load_afrag(a0, Qs, ln,      QS_STRIDE, d0, half);
      load_afrag(a1, Qs, 16 + ln, QS_STRIDE, d0, half);
      load_bfrag_nk(b, kcur, wave * 16 + ln, KS_STRIDE, half);
      sacc0 = wmma_bf16(a0, b, sacc0);
      sacc1 = wmma_bf16(a1, b, sacc1);
      wait_async();
      __syncthreads();
    }
    // Sb aliases K buffers; last loop barrier separates reads from writes
    #pragma unroll
    for (int r = 0; r < 8; ++r) {
      Sb[(     half * 8 + r) * SB_STRIDE + wave * 16 + ln] = sacc0[r] * SCALE;
      Sb[(16 + half * 8 + r) * SB_STRIDE + wave * 16 + ln] = sacc1[r] * SCALE;
    }
    __syncthreads();

    // ---- online softmax (one query row per thread, wave 0) ----
    if (tid < 32) {
      float mold = m_i[tid], lold = l_i[tid];
      float mc = -1e30f;
      for (int c = 0; c < 128; ++c) mc = fmaxf(mc, Sb[tid * SB_STRIDE + c]);
      float mnew = fmaxf(mold, mc);
      float sum = 0.f;
      for (int c = 0; c < 128; ++c) {
        float p = __expf(Sb[tid * SB_STRIDE + c] - mnew);
        Pb[tid * PB_STRIDE + c] = f32_to_bf16(p);
        sum += p;
      }
      float rs = __expf(mold - mnew);
      m_i[tid] = mnew;
      l_i[tid] = lold * rs + sum;
      rsc[tid] = rs;
    }
    __syncthreads();

    // rescale running context accumulators
    #pragma unroll
    for (int r = 0; r < 8; ++r) {
      float rv0 = rsc[     half * 8 + r];
      float rv1 = rsc[16 + half * 8 + r];
      #pragma unroll
      for (int f = 0; f < 6; ++f) { acc0[f][r] *= rv0; acc1[f][r] *= rv1; }
    }

    // ---- ctx += P.V ; keys in 32-wide slices ----
    for (int k0 = 0; k0 < 128; k0 += 32) {
      __syncthreads();                      // prior Vs readers done
      #pragma unroll
      for (int it = 0; it < 12; ++it) {
        int id = tid + it * 256;
        int r = id / 96, c = id - r * 96;
        async_b128(Vs + r * VS_STRIDE + c * 8,
                   vg + (krow0 + j + k0 + r) * DD + c * 8);
      }
      wait_async();
      __syncthreads();
      Frag16 aP0, aP1, bv0, bv1;
      load_afrag(aP0, Pb, ln,      PB_STRIDE, k0, half);
      load_afrag(aP1, Pb, 16 + ln, PB_STRIDE, k0, half);
      // B fragment: [key][dcol] gather, double-buffered over f
      #pragma unroll
      for (int i = 0; i < 8; ++i) {
        int kk = half * 16 + 2 * i;
        int nc = dbase + ln;
        bv0.u[2*i]   = Vs[ kk      * VS_STRIDE + nc];
        bv0.u[2*i+1] = Vs[(kk + 1) * VS_STRIDE + nc];
      }
      #pragma unroll
      for (int f = 0; f < 6; f += 2) {
        #pragma unroll
        for (int i = 0; i < 8; ++i) {
          int kk = half * 16 + 2 * i;
          int nc = dbase + (f + 1) * 16 + ln;
          bv1.u[2*i]   = Vs[ kk      * VS_STRIDE + nc];
          bv1.u[2*i+1] = Vs[(kk + 1) * VS_STRIDE + nc];
        }
        acc0[f] = wmma_bf16(aP0, bv0, acc0[f]);
        acc1[f] = wmma_bf16(aP1, bv0, acc1[f]);
        if (f + 2 < 6) {
          #pragma unroll
          for (int i = 0; i < 8; ++i) {
            int kk = half * 16 + 2 * i;
            int nc = dbase + (f + 2) * 16 + ln;
            bv0.u[2*i]   = Vs[ kk      * VS_STRIDE + nc];
            bv0.u[2*i+1] = Vs[(kk + 1) * VS_STRIDE + nc];
          }
        }
        acc0[f + 1] = wmma_bf16(aP0, bv1, acc0[f + 1]);
        acc1[f + 1] = wmma_bf16(aP1, bv1, acc1[f + 1]);
      }
    }
  }

  __syncthreads();
  #pragma unroll
  for (int r = 0; r < 8; ++r) {
    float  inv0 = 1.f / l_i[     half * 8 + r];
    float  inv1 = 1.f / l_i[16 + half * 8 + r];
    size_t g0 = qrow0 +      half * 8 + r;
    size_t g1 = qrow0 + 16 + half * 8 + r;
    #pragma unroll
    for (int f = 0; f < 6; ++f) {
      int c = dbase + f * 16 + ln;
      ctx[g0 * DD + c] = f32_to_bf16(acc0[f][r] * inv0);
      ctx[g1 * DD + c] = f32_to_bf16(acc1[f][r] * inv1);
    }
  }
}

// =====================================================================
// Kernel 3: output projection  out[16384 x 768] = ctx(bf16)*Wout + b
// =====================================================================
__global__ __launch_bounds__(256)
void out_proj_kernel(const unsigned short* __restrict__ c_in,
                     const float* __restrict__ w,
                     const float* __restrict__ bias,
                     float* __restrict__ out)
{
  __shared__ unsigned short As[128 * TSTR];
  __shared__ unsigned short Bs[128 * TSTR];

  const int m0   = blockIdx.x * 128;
  const int n0   = blockIdx.y * 128;
  const int tid  = threadIdx.x;
  const int wave = tid >> 5;
  const int lane = tid & 31;
  const int half = lane >> 4;
  const int ln   = lane & 15;

  f32x8 acc[8];
  #pragma unroll
  for (int f = 0; f < 8; ++f) acc[f] = zero8();

  for (int k0 = 0; k0 < DD; k0 += 32) {
    __syncthreads();
    #pragma unroll
    for (int it = 0; it < 2; ++it) {        // A tile via async DMA
      int id = tid + it * 256;
      int r = id >> 2, c = id & 3;
      async_b128(&As[r * TSTR + c * 8],
                 c_in + (size_t)(m0 + r) * DD + k0 + c * 8);
    }
    #pragma unroll 4
    for (int i = tid; i < 32 * 128; i += 256) {   // B: f32->bf16 convert
      int kk = i >> 7, nn = i & 127;
      Bs[nn * TSTR + kk] = f32_to_bf16(w[(size_t)(k0 + kk) * DD + n0 + nn]);
    }
    wait_async();
    __syncthreads();

    Frag16 a, b0, b1;
    load_afrag(a, As, wave * 16 + ln, TSTR, 0, half);
    load_bfrag_nk(b0, Bs, 0 * 16 + ln, TSTR, half);
    #pragma unroll
    for (int f = 0; f < 8; f += 2) {
      load_bfrag_nk(b1, Bs, (f + 1) * 16 + ln, TSTR, half);
      acc[f] = wmma_bf16(a, b0, acc[f]);
      if (f + 2 < 8) load_bfrag_nk(b0, Bs, (f + 2) * 16 + ln, TSTR, half);
      acc[f + 1] = wmma_bf16(a, b1, acc[f + 1]);
    }
  }

  #pragma unroll
  for (int f = 0; f < 8; ++f) {
    const int   e  = n0 + f * 16 + ln;
    const float bv = bias[e];
    #pragma unroll
    for (int r = 0; r < 8; ++r) {
      const int grow = m0 + wave * 16 + half * 8 + r;
      out[(size_t)grow * DD + e] = acc[f][r] + bv;
    }
  }
}

// =====================================================================
extern "C" void kernel_launch(void* const* d_in, const int* in_sizes, int n_in,
                              void* d_out, int out_size, void* d_ws, size_t ws_size,
                              hipStream_t stream) {
  (void)in_sizes; (void)n_in; (void)out_size; (void)ws_size;

  const float* x     = (const float*)d_in[0];
  const float* w_qkv = (const float*)d_in[1];
  const float* b_qkv = (const float*)d_in[2];
  const float* w_out = (const float*)d_in[3];
  const float* b_out = (const float*)d_in[4];
  float* out = (float*)d_out;

  unsigned short* qbuf = (unsigned short*)d_ws;
  unsigned short* kbuf = qbuf + (size_t)MTOT * DD;
  unsigned short* vbuf = kbuf + (size_t)MTOT * DD;
  unsigned short* cbuf = vbuf + (size_t)MTOT * DD;

  qkv_proj_kernel<<<dim3(MTOT / 128, D3 / 128), 256, 0, stream>>>(
      x, w_qkv, b_qkv, qbuf, kbuf, vbuf);

  flash_attn_kernel<<<dim3(MTOT / 32), 256, (size_t)ATT_SMEM, stream>>>(
      qbuf, kbuf, vbuf, cbuf);

  out_proj_kernel<<<dim3(MTOT / 128, DD / 128), 256, 0, stream>>>(
      cbuf, w_out, b_out, out);
}
